// Attention_22084721836527
// MI455X (gfx1250) — matrix operand order repaired and checked by reading, not compiled, not run on hardware
//
#include <hip/hip_runtime.h>
#include <hip/hip_bf16.h>
#include <math.h>

#define Hdim 512
#define TLEN 2048
#define BATCH 64
#define MBLK 128
#define LDA 520   // f16 elems per LDS row (512 + 8 pad, keeps 16B alignment: 520*2=1040=65*16)

typedef _Float16 v4h  __attribute__((ext_vector_type(4)));
typedef _Float16 v8h  __attribute__((ext_vector_type(8)));
typedef _Float16 v16h __attribute__((ext_vector_type(16)));
typedef float    v8f  __attribute__((ext_vector_type(8)));

// ---------------------------------------------------------------- W f32 -> f16
__global__ __launch_bounds__(256) void wconvert_kernel(const float* __restrict__ W,
                                                       _Float16* __restrict__ Wh) {
    int i = blockIdx.x * 256 + threadIdx.x;            // grid 256 -> 65536 threads, 4 elems each
    for (; i < Hdim * Hdim; i += 256 * 256) Wh[i] = (_Float16)W[i];
}

// ---------------------------------------------------------------- zero context
__global__ __launch_bounds__(256) void zeroctx_kernel(float* __restrict__ ctx) {
    ctx[blockIdx.x * 256 + threadIdx.x] = 0.0f;        // grid 128 -> 32768 floats
}

// ---------------------------------------------------------------- fused scores
// scores[row] = sum_o v[o] * tanh( sum_h X[row,h]*W[o,h] + bias[o] ),  row = b*T + t
__global__ __launch_bounds__(256, 2) void scores_kernel(
    const float*    __restrict__ X,      // (B*T, H) f32
    const _Float16* __restrict__ Wh,     // (H, H) f16 row-major (o, h)
    const float*    __restrict__ bias,   // (H)
    const float*    __restrict__ v,      // (H)
    float*          __restrict__ scores) // (B*T), aliases attn region of d_out
{
    extern __shared__ _Float16 smemA[];  // MBLK x LDA f16 = 133,120 B (of 320 KB WGP LDS)

    const int tid     = threadIdx.x;
    const int lane    = tid & 31;
    const int wave    = tid >> 5;        // 0..7, each wave owns 16 rows
    const int l15     = lane & 15;
    const int halfsel = lane >> 4;       // 0: lanes 0-15, 1: lanes 16-31
    const int row0    = blockIdx.x * MBLK;

    // ---- stage A block (MBLK x H) f32 -> f16 into LDS (uniform trip count) ----
    for (int i = tid; i < MBLK * (Hdim / 4); i += 256) {
        const int r  = i >> 7;           // /128 float4 per row
        const int c4 = i & 127;
        const float4 x = ((const float4*)(X + (size_t)(row0 + r) * Hdim))[c4];
        v4h hv = { (_Float16)x.x, (_Float16)x.y, (_Float16)x.z, (_Float16)x.w };
        *(v4h*)&smemA[r * LDA + (c4 << 2)] = hv;       // ds_store_b64
    }
    __syncthreads();

    const int mrow = wave * 16 + l15;    // A-fragment row for this lane (M = l15)

    float partial[8];
#pragma unroll
    for (int r = 0; r < 8; ++r) partial[r] = 0.0f;

    for (int nt = 0; nt < Hdim / 16; ++nt) {           // 32 N-tiles of 16 output cols
        const int   o0    = nt * 16;
        const float biasv = bias[o0 + l15];            // this lane's column o = o0+l15
        const float vscal = v[o0 + l15];

        v8f c = {};
#pragma unroll
        for (int k0 = 0; k0 < Hdim; k0 += 32) {        // 16 WMMA k-steps, fully unrolled
            // A 16x32 f16: lanes 0-15 carry K {0..7,16..23}, lanes 16-31 K {8..15,24..31}
            const int kbA = k0 + halfsel * 8;
            const v8h alo = *(const v8h*)&smemA[mrow * LDA + kbA];        // ds_load_b128
            const v8h ahi = *(const v8h*)&smemA[mrow * LDA + kbA + 16];   // ds_load_b128
            const v16h a = __builtin_shufflevector(alo, ahi,
                0,1,2,3,4,5,6,7,8,9,10,11,12,13,14,15);

            // B 32x16 f16: lane n -> col N=n, lanes 0-15 K {0..7,16..23}, lanes 16-31 K {8..15,24..31}
            const int kbB = k0 + halfsel * 8;
            const _Float16* wp = Wh + (size_t)(o0 + l15) * Hdim + kbB;
            const v8h blo = *(const v8h*)wp;                              // global_load_b128 (L2-hot)
            const v8h bhi = *(const v8h*)(wp + 16);
            const v16h b = __builtin_shufflevector(blo, bhi,
                0,1,2,3,4,5,6,7,8,9,10,11,12,13,14,15);

            c = __builtin_amdgcn_wmma_f32_16x16x32_f16(
                    false, a, false, b, (short)0, c, false, false);
        }

        // C layout: lane holds column N=l15; VGPR r -> row M = r + 8*halfsel
#pragma unroll
        for (int r = 0; r < 8; ++r) {
            const float t = tanhf(c[r] + biasv);
            partial[r] = fmaf(vscal, t, partial[r]);
        }
    }

    // reduce over the 16 columns held by lanes sharing halfsel (xor masks stay in group)
#pragma unroll
    for (int r = 0; r < 8; ++r) {
        float s = partial[r];
        s += __shfl_xor(s, 1, 32);
        s += __shfl_xor(s, 2, 32);
        s += __shfl_xor(s, 4, 32);
        s += __shfl_xor(s, 8, 32);
        if (l15 == 0) {
            const int m = wave * 16 + halfsel * 8 + r;
            scores[(size_t)row0 + m] = s;
        }
    }
}

// ---------------------------------------------------------------- masked softmax
__global__ __launch_bounds__(256) void softmax_kernel(
    const unsigned char* __restrict__ mask,  // (B,T) bool, true = padded
    float* __restrict__ attn)                // (B,T): in scores, out weights
{
    __shared__ float sred[8];
    __shared__ int   scnt[8];
    const int b    = blockIdx.x;
    const int tid  = threadIdx.x;
    const int wave = tid >> 5;

    float x[8];
    int   m[8];
    float lmax = -INFINITY;
    int   lcnt = 0;
#pragma unroll
    for (int i = 0; i < 8; ++i) {
        const int t = i * 256 + tid;
        x[i] = attn[b * TLEN + t];
        m[i] = mask[b * TLEN + t];
        if (!m[i]) { lmax = fmaxf(lmax, x[i]); ++lcnt; }
    }
#pragma unroll
    for (int off = 1; off < 32; off <<= 1) {
        lmax = fmaxf(lmax, __shfl_xor(lmax, off, 32));
        lcnt += __shfl_xor(lcnt, off, 32);
    }
    if ((tid & 31) == 0) { sred[wave] = lmax; scnt[wave] = lcnt; }
    __syncthreads();
    float gmax = -INFINITY;
    int   gcnt = 0;
#pragma unroll
    for (int i = 0; i < 8; ++i) { gmax = fmaxf(gmax, sred[i]); gcnt += scnt[i]; }
    __syncthreads();

    float e[8];
    float lsum = 0.0f;
#pragma unroll
    for (int i = 0; i < 8; ++i) {
        e[i] = m[i] ? 0.0f : __expf(x[i] - gmax);
        lsum += e[i];
    }
#pragma unroll
    for (int off = 1; off < 32; off <<= 1) lsum += __shfl_xor(lsum, off, 32);
    if ((tid & 31) == 0) sred[wave] = lsum;
    __syncthreads();
    float gsum = 0.0f;
#pragma unroll
    for (int i = 0; i < 8; ++i) gsum += sred[i];

    const float inv  = (gcnt == 0) ? 0.0f : 1.0f / gsum;
    const float unif = 1.0f / (float)TLEN;
#pragma unroll
    for (int i = 0; i < 8; ++i) {
        const int t = i * 256 + tid;
        attn[b * TLEN + t] = (gcnt == 0) ? unif : e[i] * inv;
    }
}

// ---------------------------------------------------------------- context = attn . X
__global__ __launch_bounds__(256) void context_kernel(
    const float* __restrict__ X,     // (B,T,H)
    const float* __restrict__ attn,  // (B,T)
    float* __restrict__ ctx)         // (B,H), pre-zeroed
{
    const int b  = blockIdx.x;
    const int t0 = blockIdx.y * (TLEN / 8);     // 8 T-chunks of 256
    const int h0 = threadIdx.x * 2;             // 256 threads cover H=512 as float2
    const float* xb = X + (size_t)b * TLEN * Hdim;
    const float* wb = attn + b * TLEN;

    float2 acc = make_float2(0.0f, 0.0f);
#pragma unroll 4
    for (int t = t0; t < t0 + TLEN / 8; ++t) {
        const float  w = wb[t];                 // uniform -> scalar load
        const float2 xv = *(const float2*)(xb + (size_t)t * Hdim + h0);
        acc.x = fmaf(w, xv.x, acc.x);
        acc.y = fmaf(w, xv.y, acc.y);
    }
    atomicAdd(&ctx[b * Hdim + h0],     acc.x);  // global_atomic_add_f32
    atomicAdd(&ctx[b * Hdim + h0 + 1], acc.y);
}

// ---------------------------------------------------------------- launch
extern "C" void kernel_launch(void* const* d_in, const int* in_sizes, int n_in,
                              void* d_out, int out_size, void* d_ws, size_t ws_size,
                              hipStream_t stream) {
    const float*         X    = (const float*)d_in[0];         // (B,T,H) f32
    const unsigned char* mask = (const unsigned char*)d_in[1]; // (B,T) bool
    const float*         W    = (const float*)d_in[2];         // (H,H) f32
    const float*         bias = (const float*)d_in[3];         // (H)
    const float*         v    = (const float*)d_in[4];         // (1,H)

    float* out  = (float*)d_out;
    float* ctx  = out;                     // first B*H floats
    float* attn = out + BATCH * Hdim;      // next B*T floats

    _Float16* Wh = (_Float16*)d_ws;        // 512*512 f16 = 512 KB scratch

    wconvert_kernel<<<dim3(256), 256, 0, stream>>>(W, Wh);
    zeroctx_kernel<<<dim3((BATCH * Hdim) / 256), 256, 0, stream>>>(ctx);

    const size_t ldsA = (size_t)MBLK * LDA * sizeof(_Float16); // 133,120 B
    scores_kernel<<<dim3((BATCH * TLEN) / MBLK), 256, ldsA, stream>>>(X, Wh, bias, v, attn);

    softmax_kernel<<<dim3(BATCH), 256, 0, stream>>>(mask, attn);
    context_kernel<<<dim3(BATCH, 8), 256, 0, stream>>>(X, attn, ctx);
}
